// DensityAdaptiveAttention_19628000542926
// MI455X (gfx1250) — matrix operand order repaired and checked by reading, not compiled, not run on hardware
//
#include <hip/hip_runtime.h>
#include <hip/hip_bf16.h>

// ---------------- problem constants ----------------
#define BB     2
#define CC     128
#define HH     56
#define WW     56
#define NN     (HH * WW)        // 3136
#define HEADS  4
#define HD     32               // head dim
#define AGT    49               // 7x7 agent tokens
#define APAD   64               // padded agent count
#define NT16   (NN / 16)        // 196
#define NT32   (NN / 32)        // 98
#define SCALE  0.17677669529663687f   // 32^-0.5
#define LN_EPS 1e-5f

typedef __attribute__((ext_vector_type(16))) _Float16 v16h;
typedef __attribute__((ext_vector_type(8)))  float    v8f;
typedef __attribute__((ext_vector_type(4)))  float    v4f;
typedef __attribute__((ext_vector_type(4)))  unsigned int v4u;

union HFrag { v16h h; v4u u[2]; };
union HPack { _Float16 h[2]; unsigned u; };

// ---------------- WMMA helpers (layouts per ISA 7.12.2) ----------------
// A 16x32 f16 : lane (ln,hi) row m=ln ; element runs K=[8hi..8hi+7], [16+8hi..23+8hi]
// B 32x16 f16 : lane column n=ln ; element runs K as for A (kernels 3-5 use K = e + 16*hi in A and B)
// C/D 16x16 f32: element r -> M = r + 8*hi, N = ln
__device__ __forceinline__ v8f wmma16(v16h a, v16h b, v8f c) {
    return __builtin_amdgcn_wmma_f32_16x16x32_f16(false, a, false, b, (short)0, c,
                                                  false, false);
}
// contiguous 16 halves (B-fragment / row-slice)
__device__ __forceinline__ v16h ldB(const _Float16* p) {
    HFrag f;
    const v4u* q = (const v4u*)p;
    f.u[0] = q[0]; f.u[1] = q[1];
    return f.h;
}
// A-fragment: two contiguous 8-half runs
__device__ __forceinline__ v16h ldA(const _Float16* r0, const _Float16* r1) {
    HFrag f;
    f.u[0] = *(const v4u*)r0;
    f.u[1] = *(const v4u*)r1;
    return f.h;
}

// ---------------- kernel 1: QKV projection GEMM ----------------
// per batch: [384x128] x [128xN]; scatter into WMMA-friendly layouts:
//   q  [b][h][n][d]  (scale folded in)
//   kT [b][h][n][d]
//   v  [b][h][d][n]
__global__ void k_qkv_gemm(const float* __restrict__ x,
                           const float* __restrict__ w,
                           _Float16* __restrict__ q_h,
                           _Float16* __restrict__ kT_h,
                           _Float16* __restrict__ v_h) {
    int bid = blockIdx.x;
    int tn = bid % NT16; bid /= NT16;
    int to = bid % 24;   int b = bid / 24;
    int lane = threadIdx.x;
    int ln = lane & 15, hi = lane >> 4;
    int o0 = to * 16, n0 = tn * 16;

    v8f acc = {};
    for (int kk = 0; kk < CC; kk += 32) {
        // A: weight rows, two contiguous runs of 8 floats -> f16
        const float* wr = w + (size_t)(o0 + ln) * CC + kk;
        v4f r0 = *(const v4f*)(wr + 8 * hi);
        v4f r1 = *(const v4f*)(wr + 8 * hi + 4);
        v4f r2 = *(const v4f*)(wr + 16 + 8 * hi);
        v4f r3 = *(const v4f*)(wr + 16 + 8 * hi + 4);
        v16h a;
#pragma unroll
        for (int e = 0; e < 4; ++e) {
            a[e]      = (_Float16)r0[e];
            a[e + 4]  = (_Float16)r1[e];
            a[e + 8]  = (_Float16)r2[e];
            a[e + 12] = (_Float16)r3[e];
        }
        // B: x columns (strided over channel; one-shot kernel, acceptable)
        v16h bf;
#pragma unroll
        for (int e = 0; e < 16; ++e) {
            int c = kk + (e & 7) + ((e >> 3) << 4) + (hi << 3);
            bf[e] = (_Float16)x[((size_t)b * CC + c) * NN + n0 + ln];
        }
        acc = wmma16(a, bf, acc);
    }
    int n = n0 + ln;
    if (o0 < 128) {
#pragma unroll
        for (int r = 0; r < 8; ++r) {
            int ch = o0 + r + 8 * hi;
            int head = ch >> 5, d = ch & 31;
            q_h[(((size_t)b * HEADS + head) * NN + n) * HD + d] =
                (_Float16)(acc[r] * SCALE);
        }
    } else if (o0 < 256) {
#pragma unroll
        for (int r = 0; r < 8; ++r) {
            int ch = o0 - 128 + r + 8 * hi;
            int head = ch >> 5, d = ch & 31;
            kT_h[(((size_t)b * HEADS + head) * NN + n) * HD + d] = (_Float16)acc[r];
        }
    } else {
#pragma unroll
        for (int r = 0; r < 8; ++r) {
            int ch = o0 - 256 + r + 8 * hi;
            int head = ch >> 5, d = ch & 31;
            v_h[(((size_t)b * HEADS + head) * HD + d) * NN + n] = (_Float16)acc[r];
        }
    }
}

// ---------------- kernel 2: agent pooling (8x8 block mean) ----------------
// agent_h [b][h][64][32] (zero padded), agentT_h [b][h][32][64]
__global__ void k_agent_pool(const float* __restrict__ x,
                             _Float16* __restrict__ agent_h,
                             _Float16* __restrict__ agentT_h) {
    int idx = blockIdx.x * blockDim.x + threadIdx.x;
    if (idx >= BB * CC * APAD) return;
    int aij = idx % APAD; int c = (idx / APAD) % CC; int b = idx / (APAD * CC);
    float s = 0.f;
    if (aij < AGT) {
        int ai = aij / 7, aj = aij % 7;
#pragma unroll
        for (int u = 0; u < 8; ++u)
#pragma unroll
            for (int v = 0; v < 8; ++v)
                s += x[((size_t)b * CC + c) * NN + (ai * 8 + u) * WW + (aj * 8 + v)];
        s *= (1.0f / 64.0f);
    }
    int head = c / HD, d = c % HD;
    size_t bh = (size_t)b * HEADS + head;
    agent_h[(bh * APAD + aij) * HD + d]  = (_Float16)s;
    agentT_h[(bh * HD + d) * APAD + aij] = (_Float16)s;
}

// ---------------- kernel 3: stage 1 (32 queries/wave) ----------------
__global__ void k_stage1(const _Float16* __restrict__ q_h,
                         const _Float16* __restrict__ agent_h,
                         const _Float16* __restrict__ agentT_h,
                         _Float16* __restrict__ qcomp_h) {
    __shared__ float               Sld[32][APAD + 1];
    __shared__ alignas(16) _Float16 Pld[32][APAD + 8];
    int bid = blockIdx.x;
    int tq = bid % NT32; bid /= NT32;
    int head = bid % HEADS; int b = bid / HEADS;
    int lane = threadIdx.x;
    int ln = lane & 15, hi = lane >> 4;
    int n0 = tq * 32;
    size_t bh = (size_t)b * HEADS + head;

    const _Float16* qb  = q_h + bh * NN * HD;
    const _Float16* ab  = agent_h + bh * APAD * HD;
    const _Float16* abT = agentT_h + bh * HD * APAD;

    // two A fragments of pre-scaled queries
    const _Float16* qr0 = qb + (size_t)(n0 + ln) * HD;
    const _Float16* qr1 = qb + (size_t)(n0 + 16 + ln) * HD;
    v16h aq0 = ldA(qr0 + 16 * hi, qr0 + 16 * hi + 8);
    v16h aq1 = ldA(qr1 + 16 * hi, qr1 + 16 * hi + 8);

    // scores vs 64 padded agents
#pragma unroll
    for (int at = 0; at < 4; ++at) {
        v16h bf = ldB(ab + (size_t)(at * 16 + ln) * HD + 16 * hi);
        v8f s0 = {}; s0 = wmma16(aq0, bf, s0);
        v8f s1 = {}; s1 = wmma16(aq1, bf, s1);
#pragma unroll
        for (int r = 0; r < 8; ++r) {
            Sld[r + 8 * hi][at * 16 + ln]      = s0[r];
            Sld[16 + r + 8 * hi][at * 16 + ln] = s1[r];
        }
    }
    __syncthreads();

    // softmax over 49 agents; one row per lane
    {
        float mx = -1e30f;
        for (int j = 0; j < AGT; ++j) mx = fmaxf(mx, Sld[lane][j]);
        float sum = 0.f;
        float pr[APAD];
        for (int j = 0; j < APAD; ++j) {
            float p = (j < AGT) ? __expf(Sld[lane][j] - mx) : 0.f;
            sum += p; pr[j] = p;
        }
        float inv = 1.0f / sum;
        for (int j = 0; j < APAD; ++j) Pld[lane][j] = (_Float16)(pr[j] * inv);
    }
    __syncthreads();

    // q_comp = P[32x64] @ agent[64x32]
    v8f o00 = {}, o01 = {}, o10 = {}, o11 = {};
#pragma unroll
    for (int kc = 0; kc < 2; ++kc) {
        v16h ap0 = ldA(&Pld[ln][kc * 32 + 16 * hi], &Pld[ln][kc * 32 + 16 * hi + 8]);
        v16h ap1 = ldA(&Pld[16 + ln][kc * 32 + 16 * hi],
                       &Pld[16 + ln][kc * 32 + 16 * hi + 8]);
#pragma unroll
        for (int nt = 0; nt < 2; ++nt) {
            v16h bf = ldB(abT + (size_t)(nt * 16 + ln) * APAD + kc * 32 + 16 * hi);
            if (nt == 0) { o00 = wmma16(ap0, bf, o00); o10 = wmma16(ap1, bf, o10); }
            else         { o01 = wmma16(ap0, bf, o01); o11 = wmma16(ap1, bf, o11); }
        }
    }
    // store q_comp pre-scaled, layout [b][h][n][d]
    _Float16* qc = qcomp_h + bh * NN * HD;
#pragma unroll
    for (int r = 0; r < 8; ++r) {
        int m0 = r + 8 * hi, m1 = 16 + r + 8 * hi;
        qc[(size_t)(n0 + m0) * HD + ln]      = (_Float16)(o00[r] * SCALE);
        qc[(size_t)(n0 + m0) * HD + 16 + ln] = (_Float16)(o01[r] * SCALE);
        qc[(size_t)(n0 + m1) * HD + ln]      = (_Float16)(o10[r] * SCALE);
        qc[(size_t)(n0 + m1) * HD + 16 + ln] = (_Float16)(o11[r] * SCALE);
    }
}

// ---------------- kernel 4: stage 2 flash attention (32 queries/wave) ----------------
#define SPAD 36   // Sld row stride: 144B (16B-aligned, bank-conflict free)
#define PPAD 40   // Pld row stride: 80B  (16B-aligned)
__global__ void k_stage2(const _Float16* __restrict__ kT_h,
                         const _Float16* __restrict__ v_h,
                         const _Float16* __restrict__ qcomp_h,
                         float* __restrict__ attn_f) {
    __shared__ alignas(16) float    Sld[32][SPAD];
    __shared__ alignas(16) _Float16 Pld[32][PPAD];
    __shared__ float cl[32];
    __shared__ float rs[32];
    int bid = blockIdx.x;
    int tq = bid % NT32; bid /= NT32;
    int head = bid % HEADS; int b = bid / HEADS;
    int lane = threadIdx.x;
    int ln = lane & 15, hi = lane >> 4;
    int n0 = tq * 32;
    size_t bh = (size_t)b * HEADS + head;

    const _Float16* qc = qcomp_h + bh * NN * HD;
    const _Float16* kb = kT_h + bh * NN * HD;
    const _Float16* vb = v_h + bh * HD * NN;

    const _Float16* qr0 = qc + (size_t)(n0 + ln) * HD;
    const _Float16* qr1 = qc + (size_t)(n0 + 16 + ln) * HD;
    v16h aq0 = ldA(qr0 + 16 * hi, qr0 + 16 * hi + 8);
    v16h aq1 = ldA(qr1 + 16 * hi, qr1 + 16 * hi + 8);

    float rmax = -1e30f, rsum = 0.f;   // row == lane
    v8f O00 = {}, O01 = {}, O10 = {}, O11 = {};

    for (int kc = 0; kc < NN; kc += 32) {
        // prefetch next chunk's K/V cachelines (global_prefetch_b8)
        if (kc + 32 < NN) {
            __builtin_prefetch(kb + (size_t)(kc + 32 + lane) * HD, 0, 3);
            __builtin_prefetch(vb + (size_t)lane * NN + kc + 32, 0, 3);
        }
        // scores: 4 WMMAs sharing two K-fragments
        v16h bk0 = ldB(kb + (size_t)(kc + ln) * HD + 16 * hi);
        v16h bk1 = ldB(kb + (size_t)(kc + 16 + ln) * HD + 16 * hi);
        v8f s00 = {}; s00 = wmma16(aq0, bk0, s00);
        v8f s01 = {}; s01 = wmma16(aq0, bk1, s01);
        v8f s10 = {}; s10 = wmma16(aq1, bk0, s10);
        v8f s11 = {}; s11 = wmma16(aq1, bk1, s11);
#pragma unroll
        for (int r = 0; r < 8; ++r) {
            Sld[r + 8 * hi][ln]           = s00[r];
            Sld[r + 8 * hi][16 + ln]      = s01[r];
            Sld[16 + r + 8 * hi][ln]      = s10[r];
            Sld[16 + r + 8 * hi][16 + ln] = s11[r];
        }
        __syncthreads();

        // online softmax: one row per lane; vectorized LDS reads, packed stores
        {
            float vals[32];
            const v4f* srow = (const v4f*)(&Sld[lane][0]);
#pragma unroll
            for (int j4 = 0; j4 < 8; ++j4) {
                v4f v = srow[j4];
#pragma unroll
                for (int e = 0; e < 4; ++e) vals[j4 * 4 + e] = v[e];
            }
            float mx = rmax;
#pragma unroll
            for (int j = 0; j < 32; ++j) mx = fmaxf(mx, vals[j]);
            float c = __expf(rmax - mx);
            float sum = 0.f;
            unsigned* prow = (unsigned*)(&Pld[lane][0]);
#pragma unroll
            for (int j2 = 0; j2 < 16; ++j2) {
                float p0 = __expf(vals[2 * j2] - mx);
                float p1 = __expf(vals[2 * j2 + 1] - mx);
                sum += p0 + p1;
                HPack pk;
                pk.h[0] = (_Float16)p0;
                pk.h[1] = (_Float16)p1;
                prow[j2] = pk.u;
            }
            rsum = rsum * c + sum;
            rmax = mx;
            cl[lane] = c;
        }
        __syncthreads();

        // rescale running output
#pragma unroll
        for (int r = 0; r < 8; ++r) {
            float c0 = cl[r + 8 * hi];
            float c1 = cl[16 + r + 8 * hi];
            O00[r] *= c0; O01[r] *= c0;
            O10[r] *= c1; O11[r] *= c1;
        }
        // P @ V: 4 WMMAs sharing two V-fragments
        v16h ap0 = ldA(&Pld[ln][16 * hi], &Pld[ln][16 * hi + 8]);
        v16h ap1 = ldA(&Pld[16 + ln][16 * hi], &Pld[16 + ln][16 * hi + 8]);
        v16h bv0 = ldB(vb + (size_t)ln * NN + kc + 16 * hi);
        v16h bv1 = ldB(vb + (size_t)(16 + ln) * NN + kc + 16 * hi);
        O00 = wmma16(ap0, bv0, O00);
        O01 = wmma16(ap0, bv1, O01);
        O10 = wmma16(ap1, bv0, O10);
        O11 = wmma16(ap1, bv1, O11);
        __syncthreads();   // protect Sld/Pld reuse
    }

    rs[lane] = rsum;
    __syncthreads();

    // normalize and store merged-head layout [b][n][c]
#pragma unroll
    for (int r = 0; r < 8; ++r) {
        int m0 = r + 8 * hi, m1 = 16 + r + 8 * hi;
        float i0 = 1.0f / rs[m0];
        float i1 = 1.0f / rs[m1];
        float* d0 = attn_f + ((size_t)b * NN + n0 + m0) * CC + head * HD;
        float* d1 = attn_f + ((size_t)b * NN + n0 + m1) * CC + head * HD;
        d0[ln] = O00[r] * i0;  d0[16 + ln] = O01[r] * i0;
        d1[ln] = O10[r] * i1;  d1[16 + ln] = O11[r] * i1;
    }
}

// ---------------- kernel 5: output projection + LayerNorm ----------------
__global__ void k_proj_ln(const float* __restrict__ attn_f,
                          const float* __restrict__ w_proj,
                          const float* __restrict__ b_proj,
                          const float* __restrict__ ln_g,
                          const float* __restrict__ ln_b,
                          float* __restrict__ out) {
    __shared__ float Old[16][CC + 1];
    int bid = blockIdx.x;
    int tq = bid % NT16; int b = bid / NT16;
    int lane = threadIdx.x;
    int ln = lane & 15, hi = lane >> 4;
    int n0 = tq * 16;

    // hoist 4 A fragments (attn rows, contiguous over C, f32 -> f16)
    const float* ar = attn_f + ((size_t)b * NN + n0 + ln) * CC;
    v16h afr[4];
#pragma unroll
    for (int kc = 0; kc < 4; ++kc) {
        v4f r0 = *(const v4f*)(ar + kc * 32 + 16 * hi);
        v4f r1 = *(const v4f*)(ar + kc * 32 + 16 * hi + 4);
        v4f r2 = *(const v4f*)(ar + kc * 32 + 16 * hi + 8);
        v4f r3 = *(const v4f*)(ar + kc * 32 + 16 * hi + 12);
#pragma unroll
        for (int e = 0; e < 4; ++e) {
            afr[kc][e]      = (_Float16)r0[e];
            afr[kc][e + 4]  = (_Float16)r1[e];
            afr[kc][e + 8]  = (_Float16)r2[e];
            afr[kc][e + 12] = (_Float16)r3[e];
        }
    }
#pragma unroll
    for (int ot = 0; ot < 8; ++ot) {
        int o = ot * 16 + ln;
        const float* wp = w_proj + (size_t)o * CC;
        v8f acc = {};
#pragma unroll
        for (int kc = 0; kc < 4; ++kc) {
            v4f r0 = *(const v4f*)(wp + kc * 32 + 16 * hi);
            v4f r1 = *(const v4f*)(wp + kc * 32 + 16 * hi + 4);
            v4f r2 = *(const v4f*)(wp + kc * 32 + 16 * hi + 8);
            v4f r3 = *(const v4f*)(wp + kc * 32 + 16 * hi + 12);
            v16h bf;
#pragma unroll
            for (int e = 0; e < 4; ++e) {
                bf[e]      = (_Float16)r0[e];
                bf[e + 4]  = (_Float16)r1[e];
                bf[e + 8]  = (_Float16)r2[e];
                bf[e + 12] = (_Float16)r3[e];
            }
            acc = wmma16(afr[kc], bf, acc);
        }
        float bias = b_proj[o];
#pragma unroll
        for (int r = 0; r < 8; ++r) Old[r + 8 * hi][o] = acc[r] + bias;
    }
    __syncthreads();

    if (lane < 16) {
        int n = n0 + lane;
        float mu = 0.f;
        for (int j = 0; j < CC; ++j) mu += Old[lane][j];
        mu *= (1.0f / CC);
        float var = 0.f;
        for (int j = 0; j < CC; ++j) {
            float d = Old[lane][j] - mu;
            var += d * d;
        }
        var *= (1.0f / CC);
        float inv = rsqrtf(var + LN_EPS);
        for (int j = 0; j < CC; ++j)
            out[((size_t)b * CC + j) * NN + n] =
                (Old[lane][j] - mu) * inv * ln_g[j] + ln_b[j];
    }
}

// ---------------- launch ----------------
extern "C" void kernel_launch(void* const* d_in, const int* in_sizes, int n_in,
                              void* d_out, int out_size, void* d_ws, size_t ws_size,
                              hipStream_t stream) {
    const float* x      = (const float*)d_in[0];
    const float* w_qkv  = (const float*)d_in[1];
    const float* w_proj = (const float*)d_in[2];
    const float* b_proj = (const float*)d_in[3];
    const float* ln_g   = (const float*)d_in[4];
    const float* ln_b   = (const float*)d_in[5];
    float* out = (float*)d_out;

    char* ws = (char*)d_ws;
    // workspace layout (256B aligned)
    _Float16* q_h      = (_Float16*)(ws);             // 1,605,632 B
    _Float16* kT_h     = (_Float16*)(ws + 1605632);   // 1,605,632 B
    _Float16* v_h      = (_Float16*)(ws + 3211264);   // 1,605,632 B
    _Float16* agent_h  = (_Float16*)(ws + 4816896);   //    32,768 B
    _Float16* agentT_h = (_Float16*)(ws + 4849664);   //    32,768 B
    _Float16* qcomp_h  = (_Float16*)(ws + 4882432);   // 1,605,632 B
    float*    attn_f   = (float*)   (ws + 6488064);   // 3,211,264 B  (total ~9.7 MB)

    k_qkv_gemm<<<dim3(BB * 24 * NT16), dim3(32), 0, stream>>>(x, w_qkv, q_h, kT_h, v_h);
    {
        int total = BB * CC * APAD;
        k_agent_pool<<<dim3((total + 127) / 128), dim3(128), 0, stream>>>(x, agent_h,
                                                                          agentT_h);
    }
    k_stage1<<<dim3(BB * HEADS * NT32), dim3(32), 0, stream>>>(q_h, agent_h, agentT_h,
                                                               qcomp_h);
    k_stage2<<<dim3(BB * HEADS * NT32), dim3(32), 0, stream>>>(kT_h, v_h, qcomp_h,
                                                               attn_f);
    k_proj_ln<<<dim3(BB * NT16), dim3(32), 0, stream>>>(attn_f, w_proj, b_proj,
                                                        ln_g, ln_b, out);
}
